// QuantumClassicalInterface_41205916238245
// MI455X (gfx1250) — compile-verified, matrix operand order checked by reading, not run on hardware
//
#include <hip/hip_runtime.h>

// ---------------------------------------------------------------------------
// QuantumClassicalInterface for MI455X (gfx1250), wave32 + WMMA bf16.
//   D = 1024, B = 4, S = 2048, BS = 8192
// All matmuls are "NT" (row-major over contraction), so one load pattern
// feeds both WMMA operands of v_wmma_f32_16x16x32_bf16.
// Round 3: LDS staging now uses CDNA5 async global->LDS DMA
// (global_load_async_to_lds_b128 + s_wait_asynccnt) instead of
// load-to-VGPR + ds_store, freeing registers and VALU slots.
// ---------------------------------------------------------------------------

typedef __attribute__((ext_vector_type(16))) __bf16 v16bf;
typedef __attribute__((ext_vector_type(8)))  float  v8f;
typedef __attribute__((ext_vector_type(4)))  int    v4i;

union ABTile { v16bf v; v4i q[2]; };

// Async DMA one 16-byte chunk from global memory into LDS (ASYNCcnt-tracked).
// Generic LDS pointers are {SHARED_BASE[63:32], lds_offset[31:0]}, so the
// low 32 bits are exactly the LDS-relative byte address the instruction wants.
__device__ __forceinline__ void async_ld_b128(void* lds_dst, const void* gsrc) {
  const unsigned loff = (unsigned)(size_t)lds_dst;
  asm volatile("global_load_async_to_lds_b128 %0, %1, off"
               :
               : "v"(loff), "v"(gsrc)
               : "memory");
}

__device__ __forceinline__ void wait_async0() {
  asm volatile("s_wait_asynccnt 0x0" ::: "memory");
}

// Load a 16x32 (MxK) bf16 WMMA operand tile from a row-major buffer
// (global or LDS). base points at (row0, k_origin); ld = row stride (elems).
// ISA 16-bit operand layout: lane l holds row (l&15); VGPR0-3 =
// K[(l>>4)*8 .. +7], VGPR4-7 = K[16+(l>>4)*8 .. +7]  -> two b128 loads.
__device__ __forceinline__ v16bf load_nt(const __bf16* base, int ld, int k0,
                                         int lane) {
  const int m  = lane & 15;
  const int kh = (lane >> 4) & 1;
  const __bf16* p = base + (size_t)m * ld + k0 + kh * 8;
  ABTile u;
  u.q[0] = *(const v4i*)p;
  u.q[1] = *(const v4i*)(p + 16);
  return u.v;
}

__device__ __forceinline__ v8f wmma_bf16(v16bf a, v16bf b, v8f c) {
  return __builtin_amdgcn_wmma_f32_16x16x32_bf16(false, a, false, b,
                                                 (short)0, c, false, false);
}

// ---------------------------------------------------------------------------
// Elementwise f32 -> bf16 conversion
// ---------------------------------------------------------------------------
__global__ void cvt_f32_bf16(const float* __restrict__ in,
                             __bf16* __restrict__ out, int n) {
  int i = blockIdx.x * blockDim.x + threadIdx.x;
  const int stride = gridDim.x * blockDim.x;
  for (; i < n; i += stride) out[i] = (__bf16)in[i];
}

// ---------------------------------------------------------------------------
// Per-batch transpose + convert: quantum [B][S][D] f32 -> Vt [B][D][S] bf16
// ---------------------------------------------------------------------------
__global__ void transpose_cvt(const float* __restrict__ in,
                              __bf16* __restrict__ out, int S, int D) {
  __shared__ __bf16 t[32][33];
  const int b  = blockIdx.z;
  const int s0 = blockIdx.x * 32;
  const int d0 = blockIdx.y * 32;
  const float* src = in + (size_t)b * S * D;
  __bf16* dst = out + (size_t)b * D * S;
  for (int i = threadIdx.y; i < 32; i += 8)
    t[i][threadIdx.x] = (__bf16)src[(size_t)(s0 + i) * D + d0 + threadIdx.x];
  __syncthreads();
  for (int i = threadIdx.y; i < 32; i += 8)
    dst[(size_t)(d0 + i) * S + s0 + threadIdx.x] = t[threadIdx.x][i];
}

// ---------------------------------------------------------------------------
// NT GEMM with bias + optional sigmoid:  C[m,n] = act( sum_k A[m,k]W[n,k]+b[n] )
// Block = 128 threads = 4 waves sharing one 16-row A block staged in LDS via
// async DMA. Wave = one 16x64 output tile (4 WMMA accumulators share A tile).
// grid = (M/16, N/256). K is a template param so LDS is statically sized.
// ---------------------------------------------------------------------------
template <typename OutT, bool SIG, int K>
__global__ void gemm_nt_bias(const __bf16* __restrict__ A, int lda,
                             const __bf16* __restrict__ W, int ldw,
                             const float* __restrict__ bias,
                             OutT* __restrict__ C, int ldc) {
  __shared__ __align__(16) __bf16 Ash[16][K + 8];
  const int tid  = threadIdx.x;
  const int lane = tid & 31;
  const int wave = tid >> 5;
  const int m0 = blockIdx.x * 16;
  const int n0 = (blockIdx.y * 4 + wave) * 64;

  // Stage the shared 16xK A row-block into LDS with async b128 DMA.
#pragma unroll 1
  for (int idx = tid; idx < 16 * (K / 8); idx += 128) {
    const int r = idx / (K / 8);
    const int c = (idx % (K / 8)) * 8;
    async_ld_b128(&Ash[r][c], A + (size_t)(m0 + r) * lda + c);
  }
  wait_async0();
  __syncthreads();

  const __bf16* B0 = W + (size_t)(n0 +  0) * ldw;
  const __bf16* B1 = W + (size_t)(n0 + 16) * ldw;
  const __bf16* B2 = W + (size_t)(n0 + 32) * ldw;
  const __bf16* B3 = W + (size_t)(n0 + 48) * ldw;

  v8f acc0 = {}, acc1 = {}, acc2 = {}, acc3 = {};
#pragma unroll 2
  for (int k0 = 0; k0 < K; k0 += 32) {
    v16bf a = load_nt(&Ash[0][0], K + 8, k0, lane);
    acc0 = wmma_bf16(a, load_nt(B0, ldw, k0, lane), acc0);
    acc1 = wmma_bf16(a, load_nt(B1, ldw, k0, lane), acc1);
    acc2 = wmma_bf16(a, load_nt(B2, ldw, k0, lane), acc2);
    acc3 = wmma_bf16(a, load_nt(B3, ldw, k0, lane), acc3);
  }

  // C/D layout: lane 0-15 -> rows 0..7 in v[0..7]; lane 16-31 -> rows 8..15.
  const int msub = (lane >> 4) * 8;
  const int nl   = lane & 15;
  v8f accs[4] = {acc0, acc1, acc2, acc3};
#pragma unroll
  for (int t = 0; t < 4; ++t) {
    const int col = n0 + t * 16 + nl;
    const float bv = bias[col];
#pragma unroll
    for (int j = 0; j < 8; ++j) {
      float v = accs[t][j] + bv;
      if (SIG) v = 1.0f / (1.0f + __expf(-v));
      C[(size_t)(m0 + msub + j) * ldc + col] = (OutT)v;
    }
  }
}

// ---------------------------------------------------------------------------
// Fused attention: per (batch, 16-row block):
//   stage:   c_proj rows (16x1024 bf16) -> LDS via async DMA
//   phase 1: S_tile = A_lds x (Qp rows)^T via WMMA -> LDS P (bf16, scaled)
//   softmax: cooperative row max / exp / sum in LDS (1/sum folded later)
//   phase 2: O = P x V  (P from LDS as A-operand, V from pre-transposed Vt)
//            out = classical + gate * O / rowsum
// Block = 128 threads (4 waves); grid = (S/16, B). LDS ~97 KB.
// ---------------------------------------------------------------------------
#define SROW 2056  // 2048 + 8 pad: keeps 16B alignment, dodges bank conflicts
#define AROW 1032  // 1024 + 8 pad

__global__ void attn_fused(const __bf16* __restrict__ Comb,      // [B*S][2048]
                           const __bf16* __restrict__ Vt,        // [B][D][S]
                           const float*  __restrict__ gate,      // [B*S][D]
                           const float*  __restrict__ classical, // [B*S][D]
                           const float*  __restrict__ temp_ptr,
                           float* __restrict__ out) {            // [B*S][D]
  __shared__ __align__(16) __bf16 P[16][SROW];
  __shared__ __align__(16) __bf16 Ash[16][AROW];
  __shared__ float red[16][8];
  __shared__ float rowmax[16];
  __shared__ float rowrcp[16];

  const int S = 2048, D = 1024;
  const int b    = blockIdx.y;
  const int bs0  = b * S + blockIdx.x * 16;
  const int tid  = threadIdx.x;
  const int lane = tid & 31;
  const int wave = tid >> 5;
  const float scale = 0.03125f / temp_ptr[0];  // 1/sqrt(1024)/temperature

  // ---- stage the 16x1024 c_proj row-block into LDS (async DMA) ----
#pragma unroll 1
  for (int idx = tid; idx < 16 * 128; idx += 128) {
    const int r = idx >> 7;
    const int c = (idx & 127) * 8;
    async_ld_b128(&Ash[r][c], Comb + (size_t)(bs0 + r) * 2048 + c);
  }
  wait_async0();
  __syncthreads();

  const int msub = (lane >> 4) * 8;
  const int nl   = lane & 15;

  // ---- phase 1: scores row-block (4 waves x 8 groups x 4 tiles) ----
#pragma unroll 1
  for (int g = 0; g < 8; ++g) {
    const int col0 = (wave * 8 + g) * 64;
    const __bf16* Bb0 = Comb + (size_t)(b * S + col0 +  0) * 2048 + 1024;
    const __bf16* Bb1 = Comb + (size_t)(b * S + col0 + 16) * 2048 + 1024;
    const __bf16* Bb2 = Comb + (size_t)(b * S + col0 + 32) * 2048 + 1024;
    const __bf16* Bb3 = Comb + (size_t)(b * S + col0 + 48) * 2048 + 1024;
    v8f acc0 = {}, acc1 = {}, acc2 = {}, acc3 = {};
#pragma unroll 2
    for (int k0 = 0; k0 < D; k0 += 32) {
      v16bf a = load_nt(&Ash[0][0], AROW, k0, lane);
      acc0 = wmma_bf16(a, load_nt(Bb0, 2048, k0, lane), acc0);
      acc1 = wmma_bf16(a, load_nt(Bb1, 2048, k0, lane), acc1);
      acc2 = wmma_bf16(a, load_nt(Bb2, 2048, k0, lane), acc2);
      acc3 = wmma_bf16(a, load_nt(Bb3, 2048, k0, lane), acc3);
    }
    v8f accs[4] = {acc0, acc1, acc2, acc3};
#pragma unroll
    for (int t = 0; t < 4; ++t)
#pragma unroll
      for (int j = 0; j < 8; ++j)
        P[msub + j][col0 + t * 16 + nl] = (__bf16)(accs[t][j] * scale);
  }
  __syncthreads();

  // ---- softmax over each of 16 rows (8 threads per row) ----
  const int r   = tid >> 3;
  const int seg = tid & 7;
  {
    float mx = -3.4e38f;
#pragma unroll 1
    for (int c = seg * 256; c < seg * 256 + 256; ++c)
      mx = fmaxf(mx, (float)P[r][c]);
    red[r][seg] = mx;
  }
  __syncthreads();
  if (seg == 0) {
    float m = red[r][0];
#pragma unroll
    for (int i = 1; i < 8; ++i) m = fmaxf(m, red[r][i]);
    rowmax[r] = m;
  }
  __syncthreads();
  {
    const float m = rowmax[r];
    float s = 0.f;
#pragma unroll 1
    for (int c = seg * 256; c < seg * 256 + 256; ++c) {
      float e = __expf((float)P[r][c] - m);
      P[r][c] = (__bf16)e;
      s += e;
    }
    red[r][seg] = s;
  }
  __syncthreads();
  if (seg == 0) {
    float s = 0.f;
#pragma unroll
    for (int i = 0; i < 8; ++i) s += red[r][i];
    rowrcp[r] = 1.0f / s;
  }
  __syncthreads();

  // ---- phase 2: O = P x V, fused epilogue (each wave covers 256 d-cols) --
  const __bf16* Vb = Vt + (size_t)b * D * S;
#pragma unroll 1
  for (int c = 0; c < 4; ++c) {
    const int d0 = (wave * 4 + c) * 64;
    v8f acc0 = {}, acc1 = {}, acc2 = {}, acc3 = {};
#pragma unroll 2
    for (int t0 = 0; t0 < S; t0 += 32) {
      // A operand (16x32 of P) straight from LDS: two ds_load_b128 per lane
      v16bf a = load_nt(&P[0][0], SROW, t0, lane);
      acc0 = wmma_bf16(a, load_nt(Vb + (size_t)(d0 +  0) * S, S, t0, lane), acc0);
      acc1 = wmma_bf16(a, load_nt(Vb + (size_t)(d0 + 16) * S, S, t0, lane), acc1);
      acc2 = wmma_bf16(a, load_nt(Vb + (size_t)(d0 + 32) * S, S, t0, lane), acc2);
      acc3 = wmma_bf16(a, load_nt(Vb + (size_t)(d0 + 48) * S, S, t0, lane), acc3);
    }
    v8f accs[4] = {acc0, acc1, acc2, acc3};
#pragma unroll
    for (int t = 0; t < 4; ++t) {
      const int col = d0 + t * 16 + nl;
#pragma unroll
      for (int j = 0; j < 8; ++j) {
        const int row = bs0 + msub + j;
        const float o = accs[t][j] * rowrcp[msub + j];
        const size_t idx = (size_t)row * D + col;
        out[idx] = classical[idx] + gate[idx] * o;
      }
    }
  }
}

// ---------------------------------------------------------------------------
// Host-side launch
// ---------------------------------------------------------------------------
extern "C" void kernel_launch(void* const* d_in, const int* in_sizes, int n_in,
                              void* d_out, int out_size, void* d_ws,
                              size_t ws_size, hipStream_t stream) {
  (void)in_sizes; (void)n_in; (void)out_size; (void)ws_size;

  const int B = 4, S = 2048, D = 1024;
  const int BS = B * S;

  const float* classical = (const float*)d_in[0];
  const float* quantum   = (const float*)d_in[1];
  const float* Wc        = (const float*)d_in[2];
  const float* bc        = (const float*)d_in[3];
  const float* Wq        = (const float*)d_in[4];
  const float* bq        = (const float*)d_in[5];
  const float* Wg        = (const float*)d_in[6];
  const float* bg        = (const float*)d_in[7];
  const float* temp      = (const float*)d_in[8];
  float* out             = (float*)d_out;

  // Workspace layout (bytes), total ~125.8 MB
  char* w = (char*)d_ws;
  __bf16* Xc   = (__bf16*)(w + 0);                    // [8192][1024] 16.78MB
  __bf16* Xq   = (__bf16*)(w + 16777216);             // [8192][1024] 16.78MB
  __bf16* Vt   = (__bf16*)(w + 33554432);             // [B][1024][2048] 16.78MB
  __bf16* Wcb  = (__bf16*)(w + 50331648);             // [1024][1024] 2.10MB
  __bf16* Wqb  = (__bf16*)(w + 52428800);             // [1024][1024] 2.10MB
  __bf16* Wgb  = (__bf16*)(w + 54525952);             // [1024][2048] 4.19MB
  __bf16* Comb = (__bf16*)(w + 58720256);             // [8192][2048] 33.55MB
  float*  Gate = (float*)(w + 92274688);              // [8192][1024] 33.55MB

  // 1) convert inputs to bf16 (+ per-batch transpose of quantum for P@V)
  cvt_f32_bf16<<<2048, 256, 0, stream>>>(classical, Xc, BS * D);
  cvt_f32_bf16<<<2048, 256, 0, stream>>>(quantum,   Xq, BS * D);
  cvt_f32_bf16<<<512,  256, 0, stream>>>(Wc, Wcb, D * D);
  cvt_f32_bf16<<<512,  256, 0, stream>>>(Wq, Wqb, D * D);
  cvt_f32_bf16<<<1024, 256, 0, stream>>>(Wg, Wgb, D * 2 * D);
  transpose_cvt<<<dim3(S / 32, D / 32, B), dim3(32, 8), 0, stream>>>(
      quantum, Vt, S, D);

  // 2) projections into Combined = [c_proj | q_proj] (bf16, ld=2048)
  gemm_nt_bias<__bf16, false, 1024><<<dim3(BS / 16, 4), 128, 0, stream>>>(
      Xc, D, Wcb, D, bc, Comb, 2 * D);
  gemm_nt_bias<__bf16, false, 1024><<<dim3(BS / 16, 4), 128, 0, stream>>>(
      Xq, D, Wqb, D, bq, Comb + D, 2 * D);

  // 3) interface gate = sigmoid(Combined @ Wg^T + bg), f32
  gemm_nt_bias<float, true, 2048><<<dim3(BS / 16, 4), 128, 0, stream>>>(
      Comb, 2 * D, Wgb, 2 * D, bg, Gate, D);

  // 4) fused similarity + softmax + attn@V + output blend
  attn_fused<<<dim3(S / 16, B), 128, 0, stream>>>(
      Comb, Vt, Gate, classical, temp, out);
}
